// KVMemoryModel_87411174408427
// MI455X (gfx1250) — matrix operand order repaired and checked by reading, not compiled, not run on hardware
//
#include <hip/hip_runtime.h>

// ---------------------------------------------------------------------------
// CDNA5 (gfx1250) WMMA helpers: V_WMMA_F32_16X16X32_F16, wave32.
// A tile: 16(M) x 32(K) f16.  B tile: 32(K) x 16(N) f16.  C/D: 16x16 f32.
// Layouts per CDNA5 ISA 7.12.2.  TDM descriptor per ISA ch. 8.
// ---------------------------------------------------------------------------
typedef __attribute__((ext_vector_type(16))) _Float16     v16h;
typedef __attribute__((ext_vector_type(8)))  float        v8f;
typedef __attribute__((ext_vector_type(4)))  unsigned int v4u;
typedef __attribute__((ext_vector_type(8)))  int          v8i;
typedef __attribute__((ext_vector_type(4)))  int          v4i;

#define DEV __device__ __forceinline__

DEV int lane_id() { return (int)(threadIdx.x & 31); }

// K offset of element e (0..15 halfs) within a 16x32 tile, given lane-half hi.
DEV int wmma_k_of(int e, int hi) {
    int vg = e >> 1, h = e & 1;
    return (vg < 4) ? (8 * hi + 2 * vg + h) : (16 + 8 * hi + 2 * (vg - 4) + h);
}

DEV v8f zero_v8f() {
    v8f c;
#pragma unroll
    for (int i = 0; i < 8; ++i) c[i] = 0.0f;
    return c;
}

DEV v8f wmma16(v16h a, v16h b, v8f c) {
    return __builtin_amdgcn_wmma_f32_16x16x32_f16(
        /*neg_a=*/false, a, /*neg_b=*/false, b,
        /*c_mod=*/(short)0, c, /*reuse_a=*/false, /*reuse_b=*/false);
}

// A (16xK) from f32 row-major [16 x ld], cols k0..k0+31, zero-pad k>=kmax.
DEV v16h load_a_f32(const float* A, int ld, int k0, int kmax) {
    int l = lane_id(), m = l & 15, hi = l >> 4;
    v16h a;
#pragma unroll
    for (int e = 0; e < 16; ++e) {
        int k = k0 + wmma_k_of(e, hi);
        a[e] = (k < kmax) ? (_Float16)A[m * ld + k] : (_Float16)0.0f;
    }
    return a;
}

// A (16xK) from f16 row-major LDS [16 x ld].
DEV v16h load_a_h(const _Float16* A, int ld, int k0) {
    int l = lane_id(), m = l & 15, hi = l >> 4;
    v16h a;
#pragma unroll
    for (int e = 0; e < 16; ++e) {
        int k = k0 + wmma_k_of(e, hi);
        a[e] = A[m * ld + k];
    }
    return a;
}

// B (KxN) from f32 row-major weight W[k*ld + n], tile (k0,n0), clamps.
DEV v16h load_b_f32(const float* W, int ld, int k0, int n0, int kmax, int nmax) {
    int l = lane_id(), n = l & 15, hi = l >> 4;
    v16h b;
#pragma unroll
    for (int e = 0; e < 16; ++e) {
        int k = k0 + wmma_k_of(e, hi);
        int nn = n0 + n;
        b[e] = (k < kmax && nn < nmax) ? (_Float16)W[k * ld + nn] : (_Float16)0.0f;
    }
    return b;
}

// B = M^T : B[k][n] = M[(n0+n)*ld + (k0+k)]  (for scores = q @ kp^T)
DEV v16h load_bT_f32(const float* M, int ld, int k0, int n0) {
    int l = lane_id(), n = l & 15, hi = l >> 4;
    v16h b;
#pragma unroll
    for (int e = 0; e < 16; ++e) {
        int k = k0 + wmma_k_of(e, hi);
        b[e] = (_Float16)M[(n0 + n) * ld + k];
    }
    return b;
}

// Store C (16x16 f32) into row-major [16 x ld] at col n0.
DEV void store_c_f32(float* D, int ld, int n0, v8f c) {
    int l = lane_id(), n = l & 15, hi = l >> 4;
#pragma unroll
    for (int v = 0; v < 8; ++v) D[(v + 8 * hi) * ld + n0 + n] = c[v];
}

// ---------------------------------------------------------------------------
// Tensor Data Mover: async 2D f32 tile load Global -> LDS.
// D# per CDNA5 ISA ch.8: group0 {count=1, lds_addr, global_addr, type=2},
// group1 {data_size=4B, tensor_dim0/1, tile_dim0/1, tensor_dim0_stride}.
// Completion tracked with TENSORcnt -> s_wait_tensorcnt 0.
// ---------------------------------------------------------------------------
DEV void tdm_load_2d_f32(unsigned lds_byte_off, const float* gptr,
                         unsigned width, unsigned height, unsigned stride_elems) {
    unsigned long long ga = (unsigned long long)(const void*)gptr;
    v4u g0;
    g0[0] = 1u;                                                  // count=1, user D#
    g0[1] = lds_byte_off;                                        // LDS byte address
    g0[2] = (unsigned)ga;                                        // global addr [31:0]
    g0[3] = (unsigned)((ga >> 32) & 0x01FFFFFFu) | (2u << 30);   // addr[56:32] | type=2
    v8i g1;
    g1[0] = (int)(2u << 16);                                     // data_size = 4 bytes
    g1[1] = (int)((width & 0xFFFFu) << 16);                      // tensor_dim0 lo16
    g1[2] = (int)(((width >> 16) & 0xFFFFu) |
                  ((height & 0xFFFFu) << 16));                   // dim0 hi | dim1 lo
    g1[3] = (int)(((height >> 16) & 0xFFFFu) |
                  ((width & 0xFFFFu) << 16));                    // dim1 hi | tile_dim0
    g1[4] = (int)(height & 0xFFFFu);                             // tile_dim1 (tile_dim2=0)
    g1[5] = (int)stride_elems;                                   // tensor_dim0_stride lo32
    g1[6] = 0;
    g1[7] = 0;
    v4i z4 = {0, 0, 0, 0};
#if __clang_major__ >= 23
    v8i z8 = {0, 0, 0, 0, 0, 0, 0, 0};
    __builtin_amdgcn_tensor_load_to_lds(g0, g1, z4, z4, z8, 0);
#else
    __builtin_amdgcn_tensor_load_to_lds(g0, g1, z4, z4, 0);
#endif
}

DEV void tdm_wait() { __builtin_amdgcn_s_wait_tensorcnt(0); }

// ---------------------------------------------------------------------------
// Constants from the reference.
// ---------------------------------------------------------------------------
#define Ec 4
#define Nc 1024
#define Fc 64
#define Hc 256
#define Lc 2
#define TOPK 16
#define W0S 30.0f
#define PI_F 3.14159265358979323846f

// Expert-kernel dynamic LDS partition (byte offsets; dynamic LDS starts at 0
// since the kernel declares no static __shared__).
#define XO_SCORES 0        // 16*1024 f32  = 65536
#define XO_MLP    65536    // 16*64  f32   = 4096
#define XO_LNQ    69632
#define XO_QMAT   73728
#define XO_ADO    77824
#define XO_ADAPT  81920
#define XO_ATTNW  86016    // 16*16 f32    = 1024
#define XO_TOPIDX 87040    // 16*16 i32    = 1024
#define XO_HA     88064    // 16*256 f16   = 8192
#define XO_HB     96256
#define XO_ADA    104448   // 16*128 f16   = 4096
#define XO_ADH    108544   // 16*64  f16   = 2048
#define XO_PET    110592   // 16*64  f32   = 4096  (TDM-staged pe tile)
#define XO_PAR    114688   // 16*64  f32   = 4096  (TDM-staged param tile)
#define XO_WA0    118784   // 128*64 f32   = 32768 (TDM-staged ad_w0)
#define XO_WA1    151552   // 64*64  f32   = 16384 (TDM-staged ad_w1)
#define XO_TOTAL  167936

// Final-kernel LDS partition.
#define FO_SP    0         // 16*64 f32    = 4096
#define FO_H1    4096      // 16*256 f16   = 8192
#define FO_H2    12288
#define FO_FW0   20480     // 64*256 f32   = 65536 (TDM-staged fw0)
#define FO_TOTAL 86016

// ---------------------------------------------------------------------------
// Kernel 1: per-row prep — pos-enc, line interp, router MLP + top-k routing.
// ---------------------------------------------------------------------------
__global__ void kvm_prep(const float* __restrict__ x, const float* __restrict__ lines,
                         const float* __restrict__ mg_w0, const float* __restrict__ mg_b0,
                         const float* __restrict__ mg_w1, const float* __restrict__ mg_b1,
                         const int* __restrict__ route_kp,
                         float* __restrict__ pe, float* __restrict__ param,
                         float* __restrict__ out_rawq,
                         float* __restrict__ rvals, int* __restrict__ ridx, int Bn) {
    int b = blockIdx.x * blockDim.x + threadIdx.x;
    if (b >= Bn) return;
    int rk = route_kp[0];

    float c0 = x[b * 4 + 0], c1 = x[b * 4 + 1], c2 = x[b * 4 + 2];
    float p  = x[b * 4 + 3] * 63.0f;
    float pf = floorf(p);
    float w  = p - pf;
    int il = (int)pf;
    int ih = (int)fminf(fmaxf(pf + 1.0f, 0.0f), 63.0f);
    for (int f = 0; f < 64; ++f) {
        float lo = lines[f * 64 + il], hv = lines[f * 64 + ih];
        param[b * 64 + f] = lo + w * (hv - lo);
    }

    pe[b * 64 + 0] = c0; pe[b * 64 + 1] = c1; pe[b * 64 + 2] = c2;
    float cc[3] = {c0, c1, c2};
    for (int d = 0; d < 3; ++d) {
        float fr = PI_F;
        for (int j = 0; j < 10; ++j) {
            float s = cc[d] * fr;
            pe[b * 64 + 3 + d * 20 + j * 2 + 0] = sinf(s);
            pe[b * 64 + 3 + d * 20 + j * 2 + 1] = cosf(s);
            fr *= 2.0f;
        }
    }
    pe[b * 64 + 63] = 0.0f;  // pad PE_DIM 63 -> 64

    float raw[Ec];
    for (int e = 0; e < Ec; ++e) raw[e] = mg_b1[e];
    for (int n = 0; n < 64; ++n) {
        float h = fmaxf(c0 * mg_w0[n] + c1 * mg_w0[64 + n] + c2 * mg_w0[128 + n] + mg_b0[n], 0.0f);
        for (int e = 0; e < Ec; ++e) raw[e] += h * mg_w1[n * Ec + e];
    }
    float mx = raw[0];
    for (int e = 1; e < Ec; ++e) mx = fmaxf(mx, raw[e]);
    float ex[Ec], s = 0.0f;
    for (int e = 0; e < Ec; ++e) { ex[e] = expf(raw[e] - mx); s += ex[e]; }
    float pr[Ec];
    for (int e = 0; e < Ec; ++e) {
        pr[e] = fmaxf(ex[e] / s, 1e-8f);
        out_rawq[b * 4 + e] = raw[e];
    }
    bool used[Ec] = {false, false, false, false};
    float vv[Ec]; int vi[Ec]; float vsum = 0.0f;
    for (int t = 0; t < rk; ++t) {
        int best = 0; float bv = -1e30f;
        for (int e = 0; e < Ec; ++e)
            if (!used[e] && pr[e] > bv) { bv = pr[e]; best = e; }
        used[best] = true; vv[t] = bv; vi[t] = best; vsum += bv;
    }
    for (int t = 0; t < rk; ++t) { rvals[b * 4 + t] = vv[t] / vsum; ridx[b * 4 + t] = vi[t]; }
}

// ---------------------------------------------------------------------------
// Kernel 2: kp = mem_k@wk, vp = mem_v@wv  (per expert, 1024x64 @ 64x64) WMMA.
// ---------------------------------------------------------------------------
__global__ void kvm_kvproj(const float* __restrict__ mem_k, const float* __restrict__ mem_v,
                           const float* __restrict__ wk, const float* __restrict__ wv,
                           float* __restrict__ kp, float* __restrict__ vp) {
    int bid   = blockIdx.x;
    int which = bid & 1;
    int tile  = (bid >> 1) & 63;
    int e     = bid >> 7;
    const float* src = (which ? mem_v : mem_k) + ((size_t)e * Nc + tile * 16) * Fc;
    const float* wgt = (which ? wv : wk) + (size_t)e * Fc * Fc;
    float* dst       = (which ? vp : kp) + ((size_t)e * Nc + tile * 16) * Fc;
    for (int nt = 0; nt < 4; ++nt) {
        v8f c = zero_v8f();
        for (int kt = 0; kt < 2; ++kt) {
            v16h a = load_a_f32(src, Fc, kt * 32, Fc);
            v16h b = load_b_f32(wgt, Fc, kt * 32, nt * 16, Fc, Fc);
            c = wmma16(a, b, c);
        }
        store_c_f32(dst, Fc, nt * 16, c);
    }
}

// ---------------------------------------------------------------------------
// Kernel 3: full expert pipeline for a 16-row batch tile, one wave32.
// grid = (B/16, E).  Dynamic LDS = XO_TOTAL (TDM-staged tiles included).
// ---------------------------------------------------------------------------
__global__ void kvm_expert(const float* __restrict__ pe, const float* __restrict__ param,
                           const float* __restrict__ enc_w0, const float* __restrict__ enc_b0,
                           const float* __restrict__ enc_wh, const float* __restrict__ enc_bh,
                           const float* __restrict__ enc_wo, const float* __restrict__ enc_bo,
                           const float* __restrict__ ln_g, const float* __restrict__ ln_b,
                           const float* __restrict__ wq,
                           const float* __restrict__ kp, const float* __restrict__ vp,
                           const float* __restrict__ ad_w0, const float* __restrict__ ad_b0,
                           const float* __restrict__ ad_w1, const float* __restrict__ ad_b1,
                           const float* __restrict__ ad_ln_g, const float* __restrict__ ad_ln_b,
                           float* __restrict__ all_out) {
    extern __shared__ char smem[];
    float* scores   = (float*)(smem + XO_SCORES);
    float* mlp      = (float*)(smem + XO_MLP);
    float* lnq      = (float*)(smem + XO_LNQ);
    float* qmat     = (float*)(smem + XO_QMAT);
    float* adO      = (float*)(smem + XO_ADO);
    float* adapted  = (float*)(smem + XO_ADAPT);
    float* attnw    = (float*)(smem + XO_ATTNW);
    int*   topidx   = (int*)(smem + XO_TOPIDX);
    _Float16* hA    = (_Float16*)(smem + XO_HA);
    _Float16* hB    = (_Float16*)(smem + XO_HB);
    _Float16* adA   = (_Float16*)(smem + XO_ADA);
    _Float16* adH   = (_Float16*)(smem + XO_ADH);
    float* peT      = (float*)(smem + XO_PET);
    float* paramT   = (float*)(smem + XO_PAR);
    float* WA0      = (float*)(smem + XO_WA0);
    float* WA1      = (float*)(smem + XO_WA1);

    int e  = blockIdx.y;
    int r0 = blockIdx.x * 16;
    int l  = lane_id(), nn = l & 15, hi = l >> 4;

    const float* A0 = ad_w0 + (size_t)e * 128 * Fc;
    const float* A1 = ad_w1 + (size_t)e * Fc * Fc;

    // --- Tensor Data Mover: stage reused tiles into LDS (async) ---
    tdm_load_2d_f32(XO_PET, pe + (size_t)r0 * 64, 64, 16, 64);     // pe tile (16x64)
    tdm_load_2d_f32(XO_PAR, param + (size_t)r0 * 64, 64, 16, 64);  // param tile (16x64)
    tdm_load_2d_f32(XO_WA0, A0, 64, 128, 64);                      // ad_w0 (128x64)
    tdm_load_2d_f32(XO_WA1, A1, 64, 64, 64);                       // ad_w1 (64x64)

    const float* W0 = enc_w0 + (size_t)e * 63 * Hc;
    const float* B0 = enc_b0 + (size_t)e * Hc;
    __builtin_prefetch(W0, 0, 1);

    tdm_wait();            // TENSORcnt == 0: staged tiles resident in LDS
    __syncthreads();

    // ---- SIREN layer 0: (16x64 pe) @ (63x256), K padded to 64 ----
    for (int nt = 0; nt < 16; ++nt) {
        v8f c = zero_v8f();
        for (int kt = 0; kt < 2; ++kt) {
            v16h a = load_a_f32(peT, 64, kt * 32, 64);
            v16h b = load_b_f32(W0, Hc, kt * 32, nt * 16, 63, Hc);
            c = wmma16(a, b, c);
        }
        float bias = B0[nt * 16 + nn];
#pragma unroll
        for (int v = 0; v < 8; ++v)
            hA[(v + 8 * hi) * Hc + nt * 16 + nn] = (_Float16)sinf(W0S * (c[v] + bias));
    }
    __syncthreads();

    // ---- SIREN hidden layers (256x256) x2 ----
    _Float16* hin = hA; _Float16* hout = hB;
    for (int layer = 0; layer < Lc; ++layer) {
        const float* Wl = enc_wh + ((size_t)(e * Lc + layer)) * Hc * Hc;
        const float* Bl = enc_bh + (size_t)(e * Lc + layer) * Hc;
        __builtin_prefetch(Wl, 0, 1);
        for (int nt = 0; nt < 16; ++nt) {
            v8f c = zero_v8f();
            for (int kt = 0; kt < 8; ++kt) {
                v16h a = load_a_h(hin, Hc, kt * 32);
                v16h b = load_b_f32(Wl, Hc, kt * 32, nt * 16, Hc, Hc);
                c = wmma16(a, b, c);
            }
            float bias = Bl[nt * 16 + nn];
#pragma unroll
            for (int v = 0; v < 8; ++v)
                hout[(v + 8 * hi) * Hc + nt * 16 + nn] = (_Float16)sinf(W0S * (c[v] + bias));
        }
        __syncthreads();
        _Float16* t = hin; hin = hout; hout = t;
    }

    // ---- mlp_feats: (16x256) @ (256x64) ----
    const float* Wo = enc_wo + (size_t)e * Hc * Fc;
    const float* Bo = enc_bo + (size_t)e * Fc;
    for (int nt = 0; nt < 4; ++nt) {
        v8f c = zero_v8f();
        for (int kt = 0; kt < 8; ++kt) {
            v16h a = load_a_h(hin, Hc, kt * 32);
            v16h b = load_b_f32(Wo, Fc, kt * 32, nt * 16, Hc, Fc);
            c = wmma16(a, b, c);
        }
#pragma unroll
        for (int v = 0; v < 8; ++v)
            mlp[(v + 8 * hi) * Fc + nt * 16 + nn] = c[v] + Bo[nt * 16 + nn];
    }
    __syncthreads();

    // ---- LayerNorm(mlp) * g + b  (one row per lane 0..15) ----
    if (l < 16) {
        float m = 0.0f;
        for (int f = 0; f < Fc; ++f) m += mlp[l * Fc + f];
        m *= (1.0f / Fc);
        float var = 0.0f;
        for (int f = 0; f < Fc; ++f) { float d = mlp[l * Fc + f] - m; var += d * d; }
        var *= (1.0f / Fc);
        float inv = rsqrtf(var + 1e-5f);
        for (int f = 0; f < Fc; ++f)
            lnq[l * Fc + f] = (mlp[l * Fc + f] - m) * inv * ln_g[f] + ln_b[f];
    }
    __syncthreads();

    // ---- q = LN @ wq  (16x64 @ 64x64) ----
    const float* Wq = wq + (size_t)e * Fc * Fc;
    for (int nt = 0; nt < 4; ++nt) {
        v8f c = zero_v8f();
        for (int kt = 0; kt < 2; ++kt) {
            v16h a = load_a_f32(lnq, Fc, kt * 32, Fc);
            v16h b = load_b_f32(Wq, Fc, kt * 32, nt * 16, Fc, Fc);
            c = wmma16(a, b, c);
        }
        store_c_f32(qmat, Fc, nt * 16, c);
    }
    __syncthreads();

    // ---- scores = q @ kp^T  (16x64 @ 64x1024) ----
    const float* KP = kp + (size_t)e * Nc * Fc;
    for (int nt = 0; nt < 64; ++nt) {
        v8f c = zero_v8f();
        for (int kt = 0; kt < 2; ++kt) {
            v16h a = load_a_f32(qmat, Fc, kt * 32, Fc);
            v16h b = load_bT_f32(KP, Fc, kt * 32, nt * 16);
            c = wmma16(a, b, c);
        }
        store_c_f32(scores, Nc, nt * 16, c);
    }
    __syncthreads();

    // ---- top-16 per row + softmax(score/8) (lanes 0..15) ----
    if (l < 16) {
        float* sr = scores + l * Nc;
        float tv[TOPK]; int ti[TOPK];
        for (int t = 0; t < TOPK; ++t) {
            float bv = -3.0e38f; int bi = 0;
            for (int n = 0; n < Nc; ++n) { float v = sr[n]; if (v > bv) { bv = v; bi = n; } }
            tv[t] = bv; ti[t] = bi; sr[bi] = -3.0e38f;
        }
        float mx = tv[0];
        for (int t = 1; t < TOPK; ++t) mx = fmaxf(mx, tv[t]);
        float s = 0.0f, wgt[TOPK];
        for (int t = 0; t < TOPK; ++t) { wgt[t] = expf((tv[t] - mx) * 0.125f); s += wgt[t]; }
        for (int t = 0; t < TOPK; ++t) { attnw[l * TOPK + t] = wgt[t] / s; topidx[l * TOPK + t] = ti[t]; }
    }
    __syncthreads();

    // ---- adapter + attention combine, one batch row at a time ----
    const float* Ab0 = ad_b0 + (size_t)e * Fc;
    const float* Ab1 = ad_b1 + (size_t)e * Fc;
    const float* Ag  = ad_ln_g + (size_t)e * Fc;
    const float* Abt = ad_ln_b + (size_t)e * Fc;
    const float* VP  = vp + (size_t)e * Nc * Fc;

    for (int r = 0; r < 16; ++r) {
        // build A = [v_top | param_feats] : 16(topk) x 128
        for (int i = l; i < 16 * 128; i += 32) {
            int kr = i >> 7, f = i & 127;
            float val = (f < 64) ? VP[topidx[r * TOPK + kr] * Fc + f]
                                 : paramT[r * 64 + (f - 64)];
            adA[i] = (_Float16)val;
        }
        __syncthreads();
        // layer 0: relu(A @ ad_w0 + b)  (16x128 @ 128x64), weights from LDS (TDM)
        for (int nt = 0; nt < 4; ++nt) {
            v8f c = zero_v8f();
            for (int kt = 0; kt < 4; ++kt) {
                v16h a = load_a_h(adA, 128, kt * 32);
                v16h b = load_b_f32(WA0, Fc, kt * 32, nt * 16, 128, Fc);
                c = wmma16(a, b, c);
            }
            float bias = Ab0[nt * 16 + nn];
#pragma unroll
            for (int v = 0; v < 8; ++v)
                adH[(v + 8 * hi) * Fc + nt * 16 + nn] = (_Float16)fmaxf(c[v] + bias, 0.0f);
        }
        __syncthreads();
        // layer 1: (16x64 @ 64x64) + b, weights from LDS (TDM)
        for (int nt = 0; nt < 4; ++nt) {
            v8f c = zero_v8f();
            for (int kt = 0; kt < 2; ++kt) {
                v16h a = load_a_h(adH, Fc, kt * 32);
                v16h b = load_b_f32(WA1, Fc, kt * 32, nt * 16, Fc, Fc);
                c = wmma16(a, b, c);
            }
#pragma unroll
            for (int v = 0; v < 8; ++v)
                adO[(v + 8 * hi) * Fc + nt * 16 + nn] = c[v] + Ab1[nt * 16 + nn];
        }
        __syncthreads();
        // adapted = v_top + LN(adO)*g + b   (lane per top-k row)
        if (l < 16) {
            float m = 0.0f;
            for (int f = 0; f < Fc; ++f) m += adO[l * Fc + f];
            m *= (1.0f / Fc);
            float var = 0.0f;
            for (int f = 0; f < Fc; ++f) { float d = adO[l * Fc + f] - m; var += d * d; }
            var *= (1.0f / Fc);
            float inv = rsqrtf(var + 1e-5f);
            for (int f = 0; f < Fc; ++f)
                adapted[l * Fc + f] = VP[topidx[r * TOPK + l] * Fc + f] +
                                      ((adO[l * Fc + f] - m) * inv * Ag[f] + Abt[f]);
        }
        __syncthreads();
        // out[f] = sum_k attn[k]*adapted[k][f] + mlp[r][f]
        for (int f = l; f < Fc; f += 32) {
            float acc = mlp[r * Fc + f];
            for (int k = 0; k < TOPK; ++k) acc += attnw[r * TOPK + k] * adapted[k * Fc + f];
            all_out[((size_t)(r0 + r) * Ec + e) * Fc + f] = acc;
        }
        __syncthreads();
    }
}

// ---------------------------------------------------------------------------
// Kernel 4: routed mixture + final MLP 64->256->256->256->1, sigmoid.
// fw0 staged into LDS with the Tensor Data Mover.
// ---------------------------------------------------------------------------
__global__ void kvm_final(const float* __restrict__ all_out,
                          const float* __restrict__ rvals, const int* __restrict__ ridx,
                          const int* __restrict__ route_kp,
                          const float* __restrict__ fw0, const float* __restrict__ fb0,
                          const float* __restrict__ fw1, const float* __restrict__ fb1,
                          const float* __restrict__ fw2, const float* __restrict__ fb2,
                          const float* __restrict__ fw3, const float* __restrict__ fb3,
                          float* __restrict__ refined) {
    extern __shared__ char smem[];
    float* sp     = (float*)(smem + FO_SP);
    _Float16* h1  = (_Float16*)(smem + FO_H1);
    _Float16* h2  = (_Float16*)(smem + FO_H2);
    float* FW0    = (float*)(smem + FO_FW0);
    int l = lane_id(), nn = l & 15, hi = l >> 4;
    int r0 = blockIdx.x * 16;
    int rk = route_kp[0];

    tdm_load_2d_f32(FO_FW0, fw0, 256, 64, 256);  // stage fw0 (64x256) via TDM

    for (int i = l; i < 16 * 64; i += 32) {
        int r = i >> 6, f = i & 63;
        size_t b = (size_t)(r0 + r);
        float acc = 0.0f;
        for (int j = 0; j < rk; ++j)
            acc += rvals[b * 4 + j] * all_out[(b * Ec + ridx[b * 4 + j]) * Fc + f];
        sp[i] = acc;
    }
    tdm_wait();
    __syncthreads();

    for (int nt = 0; nt < 16; ++nt) {  // 64 -> 256 (B from TDM-staged LDS)
        v8f c = zero_v8f();
        for (int kt = 0; kt < 2; ++kt) {
            v16h a = load_a_f32(sp, Fc, kt * 32, Fc);
            v16h b = load_b_f32(FW0, 256, kt * 32, nt * 16, Fc, 256);
            c = wmma16(a, b, c);
        }
#pragma unroll
        for (int v = 0; v < 8; ++v)
            h1[(v + 8 * hi) * 256 + nt * 16 + nn] = (_Float16)fmaxf(c[v] + fb0[nt * 16 + nn], 0.0f);
    }
    __syncthreads();
    for (int nt = 0; nt < 16; ++nt) {  // 256 -> 256
        v8f c = zero_v8f();
        for (int kt = 0; kt < 8; ++kt) {
            v16h a = load_a_h(h1, 256, kt * 32);
            v16h b = load_b_f32(fw1, 256, kt * 32, nt * 16, 256, 256);
            c = wmma16(a, b, c);
        }
#pragma unroll
        for (int v = 0; v < 8; ++v)
            h2[(v + 8 * hi) * 256 + nt * 16 + nn] = (_Float16)fmaxf(c[v] + fb1[nt * 16 + nn], 0.0f);
    }
    __syncthreads();
    for (int nt = 0; nt < 16; ++nt) {  // 256 -> 256 (reuse h1)
        v8f c = zero_v8f();
        for (int kt = 0; kt < 8; ++kt) {
            v16h a = load_a_h(h2, 256, kt * 32);
            v16h b = load_b_f32(fw2, 256, kt * 32, nt * 16, 256, 256);
            c = wmma16(a, b, c);
        }
#pragma unroll
        for (int v = 0; v < 8; ++v)
            h1[(v + 8 * hi) * 256 + nt * 16 + nn] = (_Float16)fmaxf(c[v] + fb2[nt * 16 + nn], 0.0f);
    }
    __syncthreads();
    {   // 256 -> 1 (N padded to 16; only column 0 valid)
        v8f c = zero_v8f();
        for (int kt = 0; kt < 8; ++kt) {
            v16h a = load_a_h(h1, 256, kt * 32);
            v16h b = load_b_f32(fw3, 1, kt * 32, 0, 256, 1);
            c = wmma16(a, b, c);
        }
        if (nn == 0) {
#pragma unroll
            for (int v = 0; v < 8; ++v) {
                float z = c[v] + fb3[0];
                refined[r0 + v + 8 * hi] = 1.0f / (1.0f + expf(-z));
            }
        }
    }
}

// ---------------------------------------------------------------------------
// Host launcher
// ---------------------------------------------------------------------------
extern "C" void kernel_launch(void* const* d_in, const int* in_sizes, int n_in,
                              void* d_out, int out_size, void* d_ws, size_t ws_size,
                              hipStream_t stream) {
    (void)n_in; (void)out_size; (void)ws_size;
    const float* x       = (const float*)d_in[0];
    const float* lines   = (const float*)d_in[1];
    const float* mem_k   = (const float*)d_in[2];
    const float* mem_v   = (const float*)d_in[3];
    const float* enc_w0  = (const float*)d_in[4];
    const float* enc_b0  = (const float*)d_in[5];
    const float* enc_wh  = (const float*)d_in[6];
    const float* enc_bh  = (const float*)d_in[7];
    const float* enc_wo  = (const float*)d_in[8];
    const float* enc_bo  = (const float*)d_in[9];
    const float* ln_g    = (const float*)d_in[10];
    const float* ln_b    = (const float*)d_in[11];
    const float* wq      = (const float*)d_in[12];
    const float* wk      = (const float*)d_in[13];
    const float* wv      = (const float*)d_in[14];
    const float* ad_w0   = (const float*)d_in[15];
    const float* ad_b0   = (const float*)d_in[16];
    const float* ad_w1   = (const float*)d_in[17];
    const float* ad_b1   = (const float*)d_in[18];
    const float* ad_ln_g = (const float*)d_in[19];
    const float* ad_ln_b = (const float*)d_in[20];
    const float* mg_w0   = (const float*)d_in[21];
    const float* mg_b0   = (const float*)d_in[22];
    const float* mg_w1   = (const float*)d_in[23];
    const float* mg_b1   = (const float*)d_in[24];
    const float* fw0     = (const float*)d_in[25];
    const float* fb0     = (const float*)d_in[26];
    const float* fw1     = (const float*)d_in[27];
    const float* fb1     = (const float*)d_in[28];
    const float* fw2     = (const float*)d_in[29];
    const float* fb2     = (const float*)d_in[30];
    const float* fw3     = (const float*)d_in[31];
    const float* fb3     = (const float*)d_in[32];
    const int*   routek  = (const int*)d_in[33];

    const int Bn = in_sizes[0] / 4;  // 16384

    float* out_f       = (float*)d_out;
    float* out_refined = out_f;            // B floats
    float* out_rawq    = out_f + Bn;       // B*4 floats

    // workspace partition (floats)
    float* pe      = (float*)d_ws;
    float* param   = pe + (size_t)Bn * 64;
    float* kp      = param + (size_t)Bn * 64;
    float* vp      = kp + (size_t)Ec * Nc * Fc;
    float* all_out = vp + (size_t)Ec * Nc * Fc;
    float* rvals   = all_out + (size_t)Bn * Ec * Fc;
    int*   ridx    = (int*)(rvals + (size_t)Bn * 4);

    kvm_prep<<<(Bn + 255) / 256, 256, 0, stream>>>(
        x, lines, mg_w0, mg_b0, mg_w1, mg_b1, routek,
        pe, param, out_rawq, rvals, ridx, Bn);

    kvm_kvproj<<<Ec * 64 * 2, 32, 0, stream>>>(mem_k, mem_v, wk, wv, kp, vp);

    dim3 eg(Bn / 16, Ec);
    kvm_expert<<<eg, 32, XO_TOTAL, stream>>>(
        pe, param, enc_w0, enc_b0, enc_wh, enc_bh, enc_wo, enc_bo,
        ln_g, ln_b, wq, kp, vp, ad_w0, ad_b0, ad_w1, ad_b1, ad_ln_g, ad_ln_b,
        all_out);

    kvm_final<<<Bn / 16, 32, FO_TOTAL, stream>>>(
        all_out, rvals, ridx, routek,
        fw0, fb0, fw1, fb1, fw2, fb2, fw3, fb3, out_refined);
}
